// QuantizedBlock_79508434583579
// MI455X (gfx1250) — matrix operand order repaired and checked by reading, not compiled, not run on hardware
//
#include <hip/hip_runtime.h>

// ---------------------------------------------------------------------------
// QuantizedBlock on MI455X (gfx1250): grouped bf16-WMMA GEMM, async-LDS loads.
//
// FLOPs = 2*4096^3 = 137 GFLOP (each row uses exactly ONE quantized W): bucket
// rows by the controller's choice, one WMMA GEMM per bucket. x (bf16, 32 MB)
// and the 4 pre-quantized W variants (bf16, 128 MB) are materialized once:
// 160 MB fits entirely in the 192 MB L2, so the GEMM streams operands from L2.
// Inner loop: global_load_async_to_lds_b128 (3-stage pipeline, ASYNCcnt<=10
// keeps one stage in flight) -> ds_load_b128 -> 8x v_wmma per K=32 step.
// ---------------------------------------------------------------------------

typedef __attribute__((ext_vector_type(16))) __bf16 v16bf;
typedef __attribute__((ext_vector_type(8)))  __bf16 v8bf;
typedef __attribute__((ext_vector_type(8)))  float  v8f;

#define B_ROWS  4096
#define IN_DIM  4096
#define OUT_DIM 4096

// GEMM tiling: block 256 threads = 8 waves (2M x 4N), wave tile 32x64.
#define BM 64
#define BN 256
#define KSTAGE 64                    // K elements per LDS stage
#define NSTAGES (IN_DIM / KSTAGE)    // 64
#define LDSROW 72                    // bf16/row: 144 B = 36 dwords, pad for banks
#define A_ELEMS (BM * LDSROW)        // 4608 bf16
#define STAGE_ELEMS ((BM + BN) * LDSROW)     // 23040 bf16
#define STAGE_BYTES (STAGE_ELEMS * 2)        // 46080 B

// ---------------- init: zero atomically-updated ws fields --------------------
__global__ void qb_init_kernel(unsigned* absmax, int* counts) {
    if (threadIdx.x == 0) *absmax = 0u;
    if (threadIdx.x < 4)  counts[threadIdx.x] = 0;
}

// ---------------- absmax over W (float max via uint atomicMax) ---------------
__global__ void qb_absmax_kernel(const float* __restrict__ W, unsigned* absmax) {
    __shared__ float red[256];
    int tid = threadIdx.x;
    size_t idx    = (size_t)blockIdx.x * blockDim.x + tid;
    size_t stride = (size_t)gridDim.x * blockDim.x;
    float m = 0.0f;
    for (size_t i = idx; i < (size_t)OUT_DIM * IN_DIM; i += stride)
        m = fmaxf(m, fabsf(W[i]));
    red[tid] = m;
    __syncthreads();
    for (int s = 128; s > 0; s >>= 1) {
        if (tid < s) red[tid] = fmaxf(red[tid], red[tid + s]);
        __syncthreads();
    }
    if (tid == 0) atomicMax(absmax, __float_as_uint(red[0]));
}

// ------ per-row stats + controller MLP + argmax + bucketing + x -> bf16 ------
__global__ void qb_stats_kernel(const float* __restrict__ x,
                                const float* __restrict__ cw1, const float* __restrict__ cb1,
                                const float* __restrict__ cw2, const float* __restrict__ cb2,
                                int* counts, int* row_list, float* bits_out,
                                __bf16* __restrict__ xbf) {
    __shared__ float s_sum[256], s_sq[256];
    __shared__ int   s_z[256];
    int row = blockIdx.x, tid = threadIdx.x;
    const float* xr = x + (size_t)row * IN_DIM;
    __bf16*      xb = xbf + (size_t)row * IN_DIM;
    float sum = 0.0f, sq = 0.0f; int z = 0;
    for (int i = tid * 2; i < IN_DIM; i += 512) {          // 8 iters, paired
        float2 v = *(const float2*)(xr + i);
        sum += v.x + v.y;
        sq  += v.x * v.x + v.y * v.y;
        z   += (v.x == 0.0f) + (v.y == 0.0f);
        union { __bf16 h[2]; unsigned u; } p;
        p.h[0] = (__bf16)v.x; p.h[1] = (__bf16)v.y;
        *(unsigned*)(xb + i) = p.u;                        // bf16 side copy
    }
    s_sum[tid] = sum; s_sq[tid] = sq; s_z[tid] = z;
    __syncthreads();
    for (int s = 128; s > 0; s >>= 1) {
        if (tid < s) {
            s_sum[tid] += s_sum[tid + s];
            s_sq[tid]  += s_sq[tid + s];
            s_z[tid]   += s_z[tid + s];
        }
        __syncthreads();
    }
    if (tid == 0) {
        const float n = (float)IN_DIM;
        float mean = s_sum[0] / n;
        float var  = (s_sq[0] - n * mean * mean) / (n - 1.0f);  // ddof=1
        float zf   = (float)s_z[0] / n;
        float h[16];
        #pragma unroll
        for (int j = 0; j < 16; j++) {
            float t = cw1[j*3+0]*mean + cw1[j*3+1]*var + cw1[j*3+2]*zf + cb1[j];
            h[j] = t > 0.0f ? t : 0.0f;
        }
        int best = 0; float bestv = -__builtin_inff();
        #pragma unroll
        for (int c = 0; c < 4; c++) {
            float l = cb2[c];
            #pragma unroll
            for (int j = 0; j < 16; j++) l += cw2[c*16+j] * h[j];
            if (l > bestv) { bestv = l; best = c; }   // first-max == jnp.argmax
        }
        const int BITS[4] = {4, 8, 16, 32};
        bits_out[row] = (float)BITS[best];
        int pos = atomicAdd(&counts[best], 1);
        row_list[best * B_ROWS + pos] = row;
    }
}

// --------- quantize W once per launch into 4 bf16 variants (L2-resident) -----
__global__ void qb_quant_kernel(const float* __restrict__ W,
                                const unsigned* __restrict__ absmax_bits,
                                __bf16* __restrict__ wq) {          // [4][4096*4096]
    const float amax = __uint_as_float(*absmax_bits);
    const float qlv[3] = {15.0f, 255.0f, 65535.0f};
    size_t i = ((size_t)blockIdx.x * blockDim.x + threadIdx.x) * 4;
    float4 v = *(const float4*)(W + i);
    union { __bf16 h[4]; uint2 u; } p;
    #pragma unroll
    for (int g = 0; g < 3; g++) {
        float scale = 2.0f * amax / (qlv[g] + 1e-9f);
        float inv_s = 1.0f / (scale + 1e-9f);
        p.h[0] = (__bf16)(__builtin_rintf(v.x * inv_s) * scale);   // RNE == jnp.round
        p.h[1] = (__bf16)(__builtin_rintf(v.y * inv_s) * scale);
        p.h[2] = (__bf16)(__builtin_rintf(v.z * inv_s) * scale);
        p.h[3] = (__bf16)(__builtin_rintf(v.w * inv_s) * scale);
        *(uint2*)(wq + (size_t)g * OUT_DIM * IN_DIM + i) = p.u;
    }
    p.h[0] = (__bf16)v.x; p.h[1] = (__bf16)v.y;                    // bits==32
    p.h[2] = (__bf16)v.z; p.h[3] = (__bf16)v.w;
    *(uint2*)(wq + (size_t)3 * OUT_DIM * IN_DIM + i) = p.u;
}

// ---------------- grouped GEMM: C = gather(xbf) @ wq_g^T ---------------------
__global__ __launch_bounds__(256)
void qb_gemm_kernel(const __bf16* __restrict__ xbf, const __bf16* __restrict__ wq,
                    const float* __restrict__ bias,
                    const float* __restrict__ gamma, const float* __restrict__ beta,
                    const float* __restrict__ run_mean, const float* __restrict__ run_var,
                    const int* __restrict__ counts, const int* __restrict__ row_list,
                    float* __restrict__ out) {
    const int g      = blockIdx.z;
    const int cnt    = counts[g];
    const int m_base = blockIdx.y * BM;
    if (m_base >= cnt) return;           // uniform early-out for empty tiles
    const int n_base = blockIdx.x * BN;
    const __bf16* wqg = wq + (size_t)g * OUT_DIM * IN_DIM;

    __shared__ __bf16 lds[3][STAGE_ELEMS];       // 3-stage async pipeline, 135 KB
    __shared__ int    rows_s[BM];

    const int tid = threadIdx.x;
    if (tid < BM) {
        int r = m_base + tid;
        rows_s[tid] = (r < cnt) ? row_list[g * B_ROWS + r] : -1;
    }
    __syncthreads();

    // -------- async-copy job for this thread: 2 A chunks + 8 B chunks --------
    // chunk = 16 B (8 bf16). id = tid + 256*c -> row = (tid>>3)+32c, chk = tid&7.
    const int r0  = tid >> 3;
    const int chk = tid & 7;
    const unsigned aoff0 = (unsigned)(r0 * (LDSROW * 2) + chk * 16);
    const unsigned boff0 = (unsigned)(A_ELEMS * 2) + aoff0;
    const unsigned lbase = (unsigned)(uintptr_t)(&lds[0][0]);   // LDS byte offset

    int ra0 = rows_s[r0];       int ra1 = rows_s[r0 + 32];
    unsigned long long aA0 = (unsigned long long)(uintptr_t)
        (xbf + (size_t)(ra0 < 0 ? 0 : ra0) * IN_DIM) + chk * 16;
    unsigned long long aA1 = (unsigned long long)(uintptr_t)
        (xbf + (size_t)(ra1 < 0 ? 0 : ra1) * IN_DIM) + chk * 16;
    unsigned long long aB  = (unsigned long long)(uintptr_t)
        (wqg + (size_t)(n_base + r0) * IN_DIM) + chk * 16;

    auto issue_stage = [&](unsigned sb, unsigned long long a0,
                           unsigned long long a1, unsigned long long b) {
        asm volatile("global_load_async_to_lds_b128 %0, %1, off"
                     :: "v"(lbase + sb + aoff0),        "v"(a0) : "memory");
        asm volatile("global_load_async_to_lds_b128 %0, %1, off"
                     :: "v"(lbase + sb + aoff0 + 4608), "v"(a1) : "memory");
        #pragma unroll
        for (int c = 0; c < 8; c++) {                       // 32-row steps of B
            asm volatile("global_load_async_to_lds_b128 %0, %1, off"
                         :: "v"(lbase + sb + boff0 + 4608u * c),
                            "v"(b + (unsigned long long)c * (32ull * IN_DIM * 2))
                         : "memory");
        }
    };

    const int wave = tid >> 5, lane = tid & 31;
    const int wm = wave & 1;            // 32-row half of the 64-row M tile
    const int wn = wave >> 1;           // 64-col quarter of the 256-col N tile
    const int half = lane >> 4, l15 = lane & 15;

    v8f acc[2][4] = {};

    // prologue: issue stages 0 and 1; wait for stage 0 only (<=10 outstanding)
    issue_stage(0u, aA0, aA1, aB);
    aA0 += KSTAGE * 2; aA1 += KSTAGE * 2; aB += KSTAGE * 2;
    issue_stage((unsigned)STAGE_BYTES, aA0, aA1, aB);
    aA0 += KSTAGE * 2; aA1 += KSTAGE * 2; aB += KSTAGE * 2;
    asm volatile("s_wait_asynccnt 0xa" ::: "memory");
    __syncthreads();

    int ci = 0;   // compute buffer (stage s % 3)
    int ii = 2;   // issue buffer   (stage s+2 % 3)
    for (int s = 0; s < NSTAGES; ++s) {
        const bool pump = (s + 2 < NSTAGES);
        if (pump) {                     // overlap stage s+2 DMA with WMMA on s
            issue_stage((unsigned)(ii * STAGE_BYTES), aA0, aA1, aB);
            aA0 += KSTAGE * 2; aA1 += KSTAGE * 2; aB += KSTAGE * 2;
            if (++ii == 3) ii = 0;
        }
        const __bf16* abuf = &lds[ci][0];
        const __bf16* bbuf = abuf + A_ELEMS;
        if (++ci == 3) ci = 0;

        #pragma unroll
        for (int kk = 0; kk < KSTAGE; kk += 32) {
            union { v16bf v; v8bf h[2]; } af[2], bf[4];
            #pragma unroll
            for (int t = 0; t < 2; t++) {   // A 16x32: half-interleaved K chunks
                const __bf16* p = abuf + (wm * 32 + t * 16 + l15) * LDSROW + kk;
                af[t].h[0] = *(const v8bf*)(p + 8 * half);
                af[t].h[1] = *(const v8bf*)(p + 16 + 8 * half);
            }
            #pragma unroll
            for (int t = 0; t < 4; t++) {   // B 32x16: half-interleaved K chunks
                const __bf16* p = bbuf + (wn * 64 + t * 16 + l15) * LDSROW + kk;
                bf[t].h[0] = *(const v8bf*)(p + 8 * half);
                bf[t].h[1] = *(const v8bf*)(p + 16 + 8 * half);
            }
            #pragma unroll
            for (int tm = 0; tm < 2; tm++)
                #pragma unroll
                for (int tn = 0; tn < 4; tn++)
                    acc[tm][tn] = __builtin_amdgcn_wmma_f32_16x16x32_bf16(
                        false, af[tm].v, false, bf[tn].v, (short)0,
                        acc[tm][tn], false, false);
        }
        // in-order ASYNCcnt: <=10 guarantees stage s+1 landed, s+2 in flight;
        // drain iterations wait to 0 so the final stages are visible.
        if (pump) asm volatile("s_wait_asynccnt 0xa" ::: "memory");
        else      asm volatile("s_wait_asynccnt 0x0" ::: "memory");
        __syncthreads();
    }

    // ---- epilogue: + bias, eval BatchNorm, ReLU, scatter by gathered row ----
    #pragma unroll
    for (int tm = 0; tm < 2; tm++) {
        #pragma unroll
        for (int tn = 0; tn < 4; tn++) {
            int n = n_base + wn * 64 + tn * 16 + l15;
            float bn_g = gamma[n], bn_b = beta[n], bn_m = run_mean[n];
            float bn_r = rsqrtf(run_var[n] + 1e-5f);
            float bv   = bias[n];
            #pragma unroll
            for (int v = 0; v < 8; v++) {
                int m    = wm * 32 + tm * 16 + v + 8 * half;  // C layout (7.12.2)
                int grow = rows_s[m];
                if (grow >= 0) {
                    float val = acc[tm][tn][v] + bv;
                    float y   = bn_g * (val - bn_m) * bn_r + bn_b;
                    y = y > 0.0f ? y : 0.0f;
                    out[(size_t)grow * OUT_DIM + n] = y;
                }
            }
        }
    }
}

// ---------------------------------------------------------------------------
extern "C" void kernel_launch(void* const* d_in, const int* in_sizes, int n_in,
                              void* d_out, int out_size, void* d_ws, size_t ws_size,
                              hipStream_t stream) {
    const float* x        = (const float*)d_in[0];
    const float* W        = (const float*)d_in[1];
    const float* b        = (const float*)d_in[2];
    const float* cw1      = (const float*)d_in[3];
    const float* cb1      = (const float*)d_in[4];
    const float* cw2      = (const float*)d_in[5];
    const float* cb2      = (const float*)d_in[6];
    const float* gamma    = (const float*)d_in[7];
    const float* beta     = (const float*)d_in[8];
    const float* run_mean = (const float*)d_in[9];
    const float* run_var  = (const float*)d_in[10];

    float* y_out    = (float*)d_out;                          // [4096, 4096]
    float* bits_out = y_out + (size_t)B_ROWS * OUT_DIM;       // [4096]

    // ws: [0] absmax | [1..4] counts | [8..] row_list[4][4096] |
    //     @128K: xbf (32 MB) | then wq[4] (128 MB). 160 MB total: L2-resident.
    char* ws = (char*)d_ws;
    unsigned* absmax = (unsigned*)ws;
    int* counts      = (int*)ws + 1;
    int* row_list    = (int*)ws + 8;
    __bf16* xbf      = (__bf16*)(ws + (128u << 10));
    __bf16* wq       = xbf + (size_t)B_ROWS * IN_DIM;

    qb_init_kernel<<<1, 64, 0, stream>>>(absmax, counts);
    qb_absmax_kernel<<<1024, 256, 0, stream>>>(W, absmax);
    qb_stats_kernel<<<B_ROWS, 256, 0, stream>>>(x, cw1, cb1, cw2, cb2,
                                                counts, row_list, bits_out, xbf);
    qb_quant_kernel<<<(OUT_DIM * IN_DIM / 4) / 256, 256, 0, stream>>>(W, absmax, wq);

    dim3 grid(OUT_DIM / BN, (B_ROWS + BM - 1) / BM, 4);
    qb_gemm_kernel<<<grid, 256, 0, stream>>>(xbf, wq, b, gamma, beta,
                                             run_mean, run_var,
                                             counts, row_list, y_out);
}
